// GCNDTI_59115929862199
// MI455X (gfx1250) — matrix operand order repaired and checked by reading, not compiled, not run on hardware
//
#include <hip/hip_runtime.h>
#include <stdint.h>

typedef __attribute__((ext_vector_type(16))) __bf16 v16bf;
typedef __attribute__((ext_vector_type(8)))  float  v8f;

#define NN 524288      // nodes
#define NE 2097152     // edges
#define NB 16384       // batch graphs
#define TPB 256

__device__ __forceinline__ unsigned short f2bf(float x) {
    unsigned int u = __float_as_uint(x);
    unsigned int r = (u + 0x7FFFu + ((u >> 16) & 1u)) >> 16;  // RNE
    return (unsigned short)r;
}
__device__ __forceinline__ float bf2f(unsigned short h) {
    return __uint_as_float(((unsigned int)h) << 16);
}

// ---------------- degree / norm ----------------
__global__ void k_deg(const int* __restrict__ dst, float* __restrict__ deg, int E) {
    int i = blockIdx.x * TPB + threadIdx.x;
    if (i < E) unsafeAtomicAdd(&deg[dst[i]], 1.0f);
}
__global__ void k_dinv(const float* __restrict__ deg, float* __restrict__ dinv, int n) {
    int i = blockIdx.x * TPB + threadIdx.x;
    if (i < n) dinv[i] = rsqrtf(deg[i] + 1.0f);   // +1 self loop; always >= 1
}

// ---------------- casts / packing ----------------
__global__ void k_cast_pad(const float* __restrict__ X, unsigned short* __restrict__ O,
                           int M, int Corig, int Cp) {
    long long idx = (long long)blockIdx.x * TPB + threadIdx.x;
    if (idx >= (long long)M * Cp) return;
    int i = (int)(idx / Cp), c = (int)(idx % Cp);
    O[idx] = (c < Corig) ? f2bf(X[(size_t)i * Corig + c]) : (unsigned short)0;
}
// W[Korig,N] f32 -> Wp[K/2, N] u32 (bf16 pair (k,k+1) per u32, zero-padded K)
__global__ void k_packw(const float* __restrict__ W, unsigned int* __restrict__ Wp,
                        int Korig, int Kp, int Nn) {
    int idx = blockIdx.x * TPB + threadIdx.x;
    int total = (Kp >> 1) * Nn;
    if (idx >= total) return;
    int kk = idx / Nn, n = idx % Nn;
    int k0 = kk * 2, k1 = k0 + 1;
    unsigned short lo = (k0 < Korig) ? f2bf(W[(size_t)k0 * Nn + n]) : (unsigned short)0;
    unsigned short hi = (k1 < Korig) ? f2bf(W[(size_t)k1 * Nn + n]) : (unsigned short)0;
    Wp[idx] = (unsigned int)lo | ((unsigned int)hi << 16);
}

// ---------------- WMMA bf16 GEMM ----------------
// One wave computes a 16x32 tile (two 16x16 accumulators sharing one A fragment).
// A bf16 [M,K] viewed as u32 [M,K/2]; Bp u32 [K/2,N]. N,K compile-time so all
// fragment loads fold to immediate offsets and the K loop unrolls into dense WMMA.
template<int Nn, int K, bool OUTF, bool RELU>
__global__ void k_gemm(const unsigned int* __restrict__ A, const unsigned int* __restrict__ Bp,
                       const float* __restrict__ bias, float* __restrict__ outF,
                       unsigned short* __restrict__ outB, int M, int ldc, int colOff) {
    constexpr int K2 = K >> 1;
    constexpr int tilesN = Nn >> 5;                 // 16x32 tiles along N
    int wave = blockIdx.x * (TPB >> 5) + ((int)threadIdx.x >> 5);
    int tiles = (M >> 4) * tilesN;
    if (wave >= tiles) return;
    int tm = wave / tilesN, tn = wave % tilesN;
    int lane = threadIdx.x & 31;
    int lh = lane >> 4;                             // half-wave select
    int ll = lane & 15;
    const unsigned int* Arow = A + (size_t)(tm * 16 + ll) * K2 + (lh << 3);
    int col0 = tn * 32 + ll;
    // B layout per lane: k = lh*16 + 2v, n = ll  ->  u32 row (lh*8 + v), col
    const unsigned int* B0 = Bp + (size_t)(lh << 3) * Nn + col0;
    v8f acc0 = {0.f, 0.f, 0.f, 0.f, 0.f, 0.f, 0.f, 0.f};
    v8f acc1 = acc0;
    #pragma unroll 4
    for (int kb = 0; kb < K2; kb += 16) {
        union { uint4 q[2]; v16bf v; } af;
        union { unsigned int u[8]; v16bf v; } b0, b1;
        // A: k = lh*16 + 2v  -> one contiguous 8-dword run
        af.q[0] = *(const uint4*)(Arow + kb);
        af.q[1] = *(const uint4*)(Arow + kb + 4);
        #pragma unroll
        for (int v = 0; v < 8; v++) {
            b0.u[v] = B0[(size_t)(kb + v) * Nn];
            b1.u[v] = B0[(size_t)(kb + v) * Nn + 16];
        }
        acc0 = __builtin_amdgcn_wmma_f32_16x16x32_bf16(false, af.v, false, b0.v,
                                                       (short)0, acc0, false, false);
        acc1 = __builtin_amdgcn_wmma_f32_16x16x32_bf16(false, af.v, false, b1.v,
                                                       (short)0, acc1, false, false);
    }
    float bv0 = bias ? bias[col0]      : 0.f;
    float bv1 = bias ? bias[col0 + 16] : 0.f;
    #pragma unroll
    for (int v = 0; v < 8; v++) {
        int m = tm * 16 + v + (lh << 3);            // C/D layout: M = v (+8 upper half)
        float x0 = acc0[v] + bv0;
        float x1 = acc1[v] + bv1;
        if (RELU) { x0 = x0 > 0.f ? x0 : 0.f; x1 = x1 > 0.f ? x1 : 0.f; }
        size_t o = (size_t)m * ldc + colOff + col0;
        if (OUTF) { outF[o] = x0; outF[o + 16] = x1; }
        else      { outB[o] = f2bf(x0); outB[o + 16] = f2bf(x1); }
    }
}

// ---------------- GCN aggregation ----------------
__global__ void k_agg(const int* __restrict__ src, const int* __restrict__ dst,
                      const float* __restrict__ dinv, const unsigned short* __restrict__ hw,
                      float* __restrict__ accum, int total) {
    int idx = blockIdx.x * TPB + threadIdx.x;
    if (idx >= total) return;
    int e = idx >> 6, f = idx & 63;
    int s = src[e], d = dst[e];
    float nrm = dinv[s] * dinv[d];
    unsafeAtomicAdd(&accum[(size_t)d * 64 + f], bf2f(hw[(size_t)s * 64 + f]) * nrm);
}
// h = relu(accum + hw*dinv^2 (self loop) + bias) -> bf16
__global__ void k_postagg(const float* __restrict__ accum, const unsigned short* __restrict__ hw,
                          const float* __restrict__ dinv, const float* __restrict__ bias,
                          unsigned short* __restrict__ hout, int total) {
    int idx = blockIdx.x * TPB + threadIdx.x;
    if (idx >= total) return;
    int i = idx >> 6, f = idx & 63;
    float di = dinv[i];
    float v = accum[idx] + bf2f(hw[idx]) * di * di + bias[f];
    hout[idx] = f2bf(v > 0.f ? v : 0.f);
}

// ---------------- pooling ----------------
__global__ void k_cnt(const int* __restrict__ batch, float* __restrict__ cnt, int n) {
    int i = blockIdx.x * TPB + threadIdx.x;
    if (i < n) unsafeAtomicAdd(&cnt[batch[i]], 1.0f);
}
__global__ void k_poolsum(const int* __restrict__ batch, const unsigned short* __restrict__ h,
                          float* __restrict__ sums, int total) {
    int idx = blockIdx.x * TPB + threadIdx.x;
    if (idx >= total) return;
    int i = idx >> 6, f = idx & 63;
    unsafeAtomicAdd(&sums[(size_t)batch[i] * 64 + f], bf2f(h[idx]));
}
__global__ void k_pooled(const float* __restrict__ sums, const float* __restrict__ cnt,
                         unsigned short* __restrict__ pooled, int total) {
    int idx = blockIdx.x * TPB + threadIdx.x;
    if (idx >= total) return;
    pooled[idx] = f2bf(sums[idx] / fmaxf(cnt[idx >> 6], 1.0f));
}

// ---------------- concat tail + layernorm ----------------
__global__ void k_copyfp(const float* __restrict__ fp, float* __restrict__ combF, int total) {
    int idx = blockIdx.x * TPB + threadIdx.x;
    if (idx >= total) return;
    int i = idx / 881, j = idx % 881;
    combF[(size_t)i * 1137 + 256 + j] = fp[idx];
}
__global__ void k_layernorm(const float* __restrict__ comb, unsigned short* __restrict__ out) {
    __shared__ float s1[TPB], s2[TPB];
    int row = blockIdx.x;
    const float* r = comb + (size_t)row * 1137;
    float a = 0.f, b = 0.f;
    for (int j = threadIdx.x; j < 1137; j += TPB) { float v = r[j]; a += v; b += v * v; }
    s1[threadIdx.x] = a; s2[threadIdx.x] = b;
    __syncthreads();
    for (int s = TPB / 2; s > 0; s >>= 1) {
        if ((int)threadIdx.x < s) { s1[threadIdx.x] += s1[threadIdx.x + s]; s2[threadIdx.x] += s2[threadIdx.x + s]; }
        __syncthreads();
    }
    float mu = s1[0] * (1.0f / 1137.0f);
    float var = s2[0] * (1.0f / 1137.0f) - mu * mu;
    float rs = rsqrtf(var + 1e-5f);
    unsigned short* o = out + (size_t)row * 1152;
    for (int j = threadIdx.x; j < 1152; j += TPB)
        o[j] = (j < 1137) ? f2bf((r[j] - mu) * rs) : (unsigned short)0;
}

// ---------------- fc3 (N=1): one wave per row ----------------
__global__ void k_fc3(const unsigned short* __restrict__ x, const float* __restrict__ w,
                      const float* __restrict__ b, float* __restrict__ out, int M) {
    int wv = blockIdx.x * (TPB >> 5) + ((int)threadIdx.x >> 5);
    int lane = threadIdx.x & 31;
    if (wv >= M) return;
    const unsigned short* xr = x + (size_t)wv * 256;
    float s = 0.f;
    #pragma unroll
    for (int j = 0; j < 8; j++) { int k = lane + 32 * j; s += bf2f(xr[k]) * w[k]; }
    #pragma unroll
    for (int o = 16; o > 0; o >>= 1) s += __shfl_xor(s, o, 32);
    if (lane == 0) out[wv] = s + b[0];
}

#define CDIV(a, b) (((a) + (b) - 1) / (b))

extern "C" void kernel_launch(void* const* d_in, const int* in_sizes, int n_in,
                              void* d_out, int out_size, void* d_ws, size_t ws_size,
                              hipStream_t stream) {
    const float* drug_x    = (const float*)d_in[0];
    const int*   edge      = (const int*)d_in[1];
    const int*   srcI      = edge;
    const int*   dstI      = edge + NE;
    const int*   batch     = (const int*)d_in[2];
    const float* protein_x = (const float*)d_in[3];
    const float* fppub     = (const float*)d_in[4];
    const float* W1 = (const float*)d_in[5],  *b1 = (const float*)d_in[6];
    const float* W2 = (const float*)d_in[7],  *b2 = (const float*)d_in[8];
    const float* W3 = (const float*)d_in[9],  *b3 = (const float*)d_in[10];
    const float* fcdw = (const float*)d_in[11], *fcdb = (const float*)d_in[12];
    const float* fcpw = (const float*)d_in[13], *fcpb = (const float*)d_in[14];
    const float* fc1w = (const float*)d_in[15], *fc1b = (const float*)d_in[16];
    const float* fc2w = (const float*)d_in[17], *fc2b = (const float*)d_in[18];
    const float* fc3w = (const float*)d_in[19], *fc3b = (const float*)d_in[20];
    float* out = (float*)d_out;

    // ---- workspace layout (256B aligned) with overlays ----
    char* ws = (char*)d_ws;
    size_t off = 0;
    auto take = [&](size_t bytes) -> char* {
        char* p = ws + off;
        off = (off + bytes + 255) & ~(size_t)255;
        return p;
    };
    float*          deg    = (float*)take((size_t)NN * 4);
    float*          dinv   = (float*)take((size_t)NN * 4);
    float*          accum  = (float*)take((size_t)NN * 64 * 4);
    unsigned short* hw     = (unsigned short*)take((size_t)NN * 64 * 2);
    unsigned short* hcur   = (unsigned short*)take((size_t)NN * 64 * 2);
    unsigned int*   W1p    = (unsigned int*)take(16 * 64 * 4);     // Kp=32
    unsigned int*   W2p    = (unsigned int*)take(32 * 64 * 4);     // Kp=64
    unsigned int*   W3p    = (unsigned int*)take(32 * 64 * 4);
    unsigned int*   fcdp   = (unsigned int*)take(32 * 128 * 4);    // Kp=64
    unsigned int*   fcpp   = (unsigned int*)take(496 * 128 * 4);   // Kp=992
    unsigned int*   fc1p   = (unsigned int*)take(576 * 512 * 4);   // Kp=1152
    unsigned int*   fc2p   = (unsigned int*)take(256 * 256 * 4);   // Kp=512
    unsigned short* protB  = (unsigned short*)take((size_t)NB * 992 * 2);
    unsigned short* pooledB= (unsigned short*)take((size_t)NB * 64 * 2);
    // overlays (regions free by the time they're reused)
    float*          sums   = accum;                                   // [NB,64]
    float*          cnt    = accum + (size_t)NB * 64;                 // [NB]
    float*          combF  = accum + (size_t)NB * 64 + NB;            // [NB,1137]
    unsigned short* combB  = hw;                                      // [NB,1152]
    unsigned short* fc1out = hw + (size_t)NB * 1152;                  // [NB,512]
    unsigned short* fc2out = fc1out + (size_t)NB * 512;               // [NB,256]

    // ---- degree + norm ----
    hipMemsetAsync(deg, 0, (size_t)NN * 4, stream);
    k_deg<<<CDIV(NE, TPB), TPB, 0, stream>>>(dstI, deg, NE);
    k_dinv<<<CDIV(NN, TPB), TPB, 0, stream>>>(deg, dinv, NN);

    // ---- weight packing (f32 -> packed bf16) ----
    k_packw<<<CDIV(16 * 64, TPB), TPB, 0, stream>>>(W1, W1p, 30, 32, 64);
    k_packw<<<CDIV(32 * 64, TPB), TPB, 0, stream>>>(W2, W2p, 64, 64, 64);
    k_packw<<<CDIV(32 * 64, TPB), TPB, 0, stream>>>(W3, W3p, 64, 64, 64);
    k_packw<<<CDIV(32 * 128, TPB), TPB, 0, stream>>>(fcdw, fcdp, 64, 64, 128);
    k_packw<<<CDIV(496 * 128, TPB), TPB, 0, stream>>>(fcpw, fcpp, 979, 992, 128);
    k_packw<<<CDIV(576 * 512, TPB), TPB, 0, stream>>>(fc1w, fc1p, 1137, 1152, 512);
    k_packw<<<CDIV(256 * 256, TPB), TPB, 0, stream>>>(fc2w, fc2p, 512, 512, 256);

    // ---- node features -> bf16 [NN,32] (stored in hcur) ----
    k_cast_pad<<<CDIV((size_t)NN * 32, TPB), TPB, 0, stream>>>(drug_x, hcur, NN, 30, 32);

    const int totNF = NN * 64;
    const int totEF = NE * 64;
    const int gcnTiles = (NN / 16) * (64 / 32);

    // ---- GCN layer 1 (K=32) ----
    k_gemm<64, 32, false, false><<<CDIV(gcnTiles, 8), TPB, 0, stream>>>(
        (const unsigned int*)hcur, W1p, nullptr, nullptr, hw, NN, 64, 0);
    hipMemsetAsync(accum, 0, (size_t)NN * 64 * 4, stream);
    k_agg<<<CDIV(totEF, TPB), TPB, 0, stream>>>(srcI, dstI, dinv, hw, accum, totEF);
    k_postagg<<<CDIV(totNF, TPB), TPB, 0, stream>>>(accum, hw, dinv, b1, hcur, totNF);

    // ---- GCN layers 2 & 3 (K=64) ----
    const float* gb[2] = { b2, b3 };
    const unsigned int* gw[2] = { W2p, W3p };
    for (int L = 0; L < 2; L++) {
        k_gemm<64, 64, false, false><<<CDIV(gcnTiles, 8), TPB, 0, stream>>>(
            (const unsigned int*)hcur, gw[L], nullptr, nullptr, hw, NN, 64, 0);
        hipMemsetAsync(accum, 0, (size_t)NN * 64 * 4, stream);
        k_agg<<<CDIV(totEF, TPB), TPB, 0, stream>>>(srcI, dstI, dinv, hw, accum, totEF);
        k_postagg<<<CDIV(totNF, TPB), TPB, 0, stream>>>(accum, hw, dinv, gb[L], hcur, totNF);
    }

    // ---- global mean pool ----
    hipMemsetAsync(accum, 0, ((size_t)NB * 64 + NB) * 4, stream);   // sums + cnt
    k_cnt<<<CDIV(NN, TPB), TPB, 0, stream>>>(batch, cnt, NN);
    k_poolsum<<<CDIV(totNF, TPB), TPB, 0, stream>>>(batch, hcur, sums, totNF);
    k_pooled<<<CDIV(NB * 64, TPB), TPB, 0, stream>>>(sums, cnt, pooledB, NB * 64);

    // ---- protein features -> bf16 [NB,992] ----
    k_cast_pad<<<CDIV((size_t)NB * 992, TPB), TPB, 0, stream>>>(protein_x, protB, NB, 979, 992);

    // ---- drug / prot FC into comb f32 [NB,1137] ----
    {
        int tiles = (NB / 16) * (128 / 32);
        k_gemm<128, 64, true, true><<<CDIV(tiles, 8), TPB, 0, stream>>>(
            (const unsigned int*)pooledB, fcdp, fcdb, combF, nullptr, NB, 1137, 0);
        k_gemm<128, 992, true, true><<<CDIV(tiles, 8), TPB, 0, stream>>>(
            (const unsigned int*)protB, fcpp, fcpb, combF, nullptr, NB, 1137, 128);
    }
    k_copyfp<<<CDIV(NB * 881, TPB), TPB, 0, stream>>>(fppub, combF, NB * 881);

    // ---- layernorm -> bf16 [NB,1152] ----
    k_layernorm<<<NB, TPB, 0, stream>>>(combF, combB);

    // ---- fc1 / fc2 (WMMA) ----
    {
        int tiles = (NB / 16) * (512 / 32);
        k_gemm<512, 1152, false, true><<<CDIV(tiles, 8), TPB, 0, stream>>>(
            (const unsigned int*)combB, fc1p, fc1b, nullptr, fc1out, NB, 512, 0);
    }
    {
        int tiles = (NB / 16) * (256 / 32);
        k_gemm<256, 512, false, true><<<CDIV(tiles, 8), TPB, 0, stream>>>(
            (const unsigned int*)fc1out, fc2p, fc2b, nullptr, fc2out, NB, 256, 0);
    }

    // ---- fc3 -> d_out [NB,1] ----
    k_fc3<<<CDIV(NB, 8), TPB, 0, stream>>>(fc2out, fc3w, fc3b, out, NB);
}
